// Speller_60859686584886
// MI455X (gfx1250) — matrix operand order repaired and checked by reading, not compiled, not run on hardware
//
#include <hip/hip_runtime.h>
#include <hip/hip_bf16.h>

typedef __bf16 bf16;
typedef __attribute__((ext_vector_type(16))) __bf16 v16bf;
typedef __attribute__((ext_vector_type(8)))  __bf16 v8bf;
typedef __attribute__((ext_vector_type(8)))  float  v8f;

#define BB   32
#define TT   256
#define DD   512
#define HH   512
#define VV   2000
#define MM   256
#define GG   2048          // 4*H
#define KP0  2528          // V+H padded to multiple of 32
#define TDEC 100

// ---------------------------------------------------------------------------
// WMMA fragment loads, contiguous k order: lane half h holds k = 16h..16h+15 for A and B (wave32)
// ---------------------------------------------------------------------------
__device__ __forceinline__ v16bf load_a_frag(const bf16* __restrict__ A, int lda,
                                             int m0, int k0, int lane) {
  int m    = m0 + (lane & 15);
  int koff = (lane >> 4) * 16;
  const bf16* p = A + (size_t)m * lda + k0 + koff;
  return *(const v16bf*)p;            // K = k0+koff .. +15
}

__device__ __forceinline__ v16bf load_b_frag(const bf16* __restrict__ W, int ldw,
                                             int n0, int k0, int lane) {
  int n    = n0 + (lane & 15);
  int koff = (lane >> 4) * 16;
  const bf16* p = W + (size_t)n * ldw + k0 + koff; // 16 contiguous bf16
  return *(const v16bf*)p;
}

// ---------------------------------------------------------------------------
// OUT[m,n] = act( sum_k A1[m,k]W1[n,k] + sum_k A2[m,k]W2[n,k] + b1[n] + b2[n] )
// Each wave: one 16-col tile x 32 rows (two WMMAs per K-chunk, shared B frag).
// blockDim = 256 (8 waves). grid.x covers ceil(N/16/8), grid.y covers M/32.
// ---------------------------------------------------------------------------
__global__ void wmma_gemm_dual(const bf16* __restrict__ A1, int lda1, int K1,
                               const bf16* __restrict__ A2, int lda2, int K2,
                               const bf16* __restrict__ W1,
                               const bf16* __restrict__ W2,
                               const float* __restrict__ bias1,
                               const float* __restrict__ bias2,
                               float* __restrict__ OUT, int N, int act) {
  int lane = threadIdx.x & 31;
  int wave = threadIdx.x >> 5;
  int tile = blockIdx.x * 8 + wave;
  if (tile * 16 >= N) return;
  int n0 = tile * 16;
  int m0 = blockIdx.y * 32;

  v8f acc0 = {};
  v8f acc1 = {};
  for (int k0 = 0; k0 < K1; k0 += 32) {
    v16bf b  = load_b_frag(W1, K1, n0, k0, lane);
    v16bf a0 = load_a_frag(A1, lda1, m0,      k0, lane);
    v16bf a1 = load_a_frag(A1, lda1, m0 + 16, k0, lane);
    acc0 = __builtin_amdgcn_wmma_f32_16x16x32_bf16(false, a0, false, b, (short)0, acc0, false, false);
    acc1 = __builtin_amdgcn_wmma_f32_16x16x32_bf16(false, a1, false, b, (short)0, acc1, false, false);
  }
  if (A2) {
    for (int k0 = 0; k0 < K2; k0 += 32) {
      v16bf b  = load_b_frag(W2, K2, n0, k0, lane);
      v16bf a0 = load_a_frag(A2, lda2, m0,      k0, lane);
      v16bf a1 = load_a_frag(A2, lda2, m0 + 16, k0, lane);
      acc0 = __builtin_amdgcn_wmma_f32_16x16x32_bf16(false, a0, false, b, (short)0, acc0, false, false);
      acc1 = __builtin_amdgcn_wmma_f32_16x16x32_bf16(false, a1, false, b, (short)0, acc1, false, false);
    }
  }
  int n = n0 + (lane & 15);
  float bv = (bias1 ? bias1[n] : 0.f) + (bias2 ? bias2[n] : 0.f);
  int rbase = m0 + ((lane >> 4) << 3);
#pragma unroll
  for (int r = 0; r < 8; ++r) {
    float v0 = acc0[r] + bv;
    float v1 = acc1[r] + bv;
    if (act == 1) { v0 = fmaxf(v0, 0.f); v1 = fmaxf(v1, 0.f); }
    OUT[(size_t)(rbase + r)      * N + n] = v0;
    OUT[(size_t)(rbase + r + 16) * N + n] = v1;
  }
}

// ---------------------------------------------------------------------------
// elementwise / reduction kernels
// ---------------------------------------------------------------------------
__device__ __forceinline__ float sigmoidf_(float x) { return 1.f / (1.f + __expf(-x)); }

__global__ void cvt_pad_kernel(const float* __restrict__ src, bf16* __restrict__ dst,
                               int rows, int K, int Kp) {
  long long gid = (long long)blockIdx.x * blockDim.x + threadIdx.x;
  long long tot = (long long)rows * Kp;
  if (gid >= tot) return;
  int r = (int)(gid / Kp), k = (int)(gid % Kp);
  dst[gid] = (k < K) ? (bf16)src[(size_t)r * K + k] : (bf16)0.f;
}

__global__ void init_state_kernel(const float* __restrict__ listener,
                                  bf16* __restrict__ x_bf, float* __restrict__ c0,
                                  float* __restrict__ c1, bf16* __restrict__ h0_bf,
                                  bf16* __restrict__ h1_bf) {
  int gid = blockIdx.x * blockDim.x + threadIdx.x;
  if (gid < BB * HH) {
    c0[gid] = 0.f; c1[gid] = 0.f;
    h0_bf[gid] = (bf16)0.f; h1_bf[gid] = (bf16)0.f;
  }
  if (gid >= BB * KP0) return;
  int b = gid / KP0, k = gid % KP0;
  float v;
  if (k == 0)          v = 1.f;                                        // onehot0
  else if (k < VV)     v = 0.f;
  else if (k < VV + HH) v = listener[(size_t)b * TT * DD + (k - VV)];  // listener[:,0,:]
  else                 v = 0.f;                                        // K pad
  x_bf[gid] = (bf16)v;
}

__global__ void lstm_cell_kernel(const float* __restrict__ gates, // [B,4H] (i,f,g,o)
                                 float* __restrict__ c,           // [B,H]
                                 bf16* __restrict__ h_bf,         // [B,H]
                                 bf16* __restrict__ concat_bf) {  // [B,2H] or null
  int gid = blockIdx.x * blockDim.x + threadIdx.x;
  if (gid >= BB * HH) return;
  int b = gid >> 9, j = gid & (HH - 1);
  const float* g = gates + (size_t)b * GG;
  float i  = sigmoidf_(g[j]);
  float f  = sigmoidf_(g[HH + j]);
  float gg = tanhf(g[2 * HH + j]);
  float o  = sigmoidf_(g[3 * HH + j]);
  float cn = f * c[gid] + i * gg;
  c[gid] = cn;
  float h = o * tanhf(cn);
  h_bf[gid] = (bf16)h;
  if (concat_bf) concat_bf[(size_t)b * (2 * HH) + j] = (bf16)h;
}

// one block per batch element; 256 threads (== T)
__global__ void attend_kernel(const float* __restrict__ comp_dec,  // [B,M]
                              const float* __restrict__ comp_lis,  // [B*T,M]
                              const float* __restrict__ listener,  // [B,T,D]
                              float* __restrict__ attn_out,        // [B,T] (into d_out)
                              bf16* __restrict__ concat_bf,        // [B,2H]
                              bf16* __restrict__ x_bf) {           // [B,KP0]
  __shared__ float s_dec[MM];
  __shared__ float s_attn[TT];
  __shared__ float s_red[TT];
  int b = blockIdx.x, t = threadIdx.x;
  s_dec[t] = comp_dec[(size_t)b * MM + t];
  __syncthreads();
  const float* cl = comp_lis + ((size_t)b * TT + t) * MM;
  float e = 0.f;
  for (int m = 0; m < MM; ++m) e += s_dec[m] * cl[m];
  s_red[t] = e; __syncthreads();
  for (int s = 128; s > 0; s >>= 1) { if (t < s) s_red[t] = fmaxf(s_red[t], s_red[t + s]); __syncthreads(); }
  float mx = s_red[0]; __syncthreads();
  float ex = __expf(e - mx);
  s_red[t] = ex; __syncthreads();
  for (int s = 128; s > 0; s >>= 1) { if (t < s) s_red[t] += s_red[t + s]; __syncthreads(); }
  float a = ex / s_red[0];
  s_attn[t] = a;
  attn_out[(size_t)b * TT + t] = a;
  __syncthreads();
  const float* lb = listener + (size_t)b * TT * DD;
  float c0 = 0.f, c1 = 0.f;
  for (int tt = 0; tt < TT; ++tt) {
    float at = s_attn[tt];
    c0 += at * lb[(size_t)tt * DD + t];
    c1 += at * lb[(size_t)tt * DD + t + 256];
  }
  concat_bf[(size_t)b * (2 * HH) + HH + t]       = (bf16)c0;
  concat_bf[(size_t)b * (2 * HH) + HH + t + 256] = (bf16)c1;
  x_bf[(size_t)b * KP0 + VV + t]       = (bf16)c0;
  x_bf[(size_t)b * KP0 + VV + t + 256] = (bf16)c1;
}

// one block per batch element; log_softmax over V=2000 + feedback write
__global__ void logsoftmax_kernel(const float* __restrict__ logits, // [B,V]
                                  float* __restrict__ preds,        // [B,V] (into d_out)
                                  bf16* __restrict__ x_bf) {        // [B,KP0]
  __shared__ float s_red[256];
  int b = blockIdx.x, t = threadIdx.x;
  const float* lrow = logits + (size_t)b * VV;
  float mx = -1e30f;
  for (int n = t; n < VV; n += 256) mx = fmaxf(mx, lrow[n]);
  s_red[t] = mx; __syncthreads();
  for (int s = 128; s > 0; s >>= 1) { if (t < s) s_red[t] = fmaxf(s_red[t], s_red[t + s]); __syncthreads(); }
  mx = s_red[0]; __syncthreads();
  float sm = 0.f;
  for (int n = t; n < VV; n += 256) sm += __expf(lrow[n] - mx);
  s_red[t] = sm; __syncthreads();
  for (int s = 128; s > 0; s >>= 1) { if (t < s) s_red[t] += s_red[t + s]; __syncthreads(); }
  float lse = mx + __logf(s_red[0]);
  for (int n = t; n < VV; n += 256) {
    float p = lrow[n] - lse;
    preds[(size_t)b * VV + n] = p;
    x_bf[(size_t)b * KP0 + n] = (bf16)p;
  }
}

// ---------------------------------------------------------------------------
extern "C" void kernel_launch(void* const* d_in, const int* in_sizes, int n_in,
                              void* d_out, int out_size, void* d_ws, size_t ws_size,
                              hipStream_t stream) {
  const float* listener = (const float*)d_in[0];
  const float* Wih0 = (const float*)d_in[1];
  const float* Whh0 = (const float*)d_in[2];
  const float* bih0 = (const float*)d_in[3];
  const float* bhh0 = (const float*)d_in[4];
  const float* Wih1 = (const float*)d_in[5];
  const float* Whh1 = (const float*)d_in[6];
  const float* bih1 = (const float*)d_in[7];
  const float* bhh1 = (const float*)d_in[8];
  const float* Wphi = (const float*)d_in[9];
  const float* bphi = (const float*)d_in[10];
  const float* Wpsi = (const float*)d_in[11];
  const float* bpsi = (const float*)d_in[12];
  const float* Wc   = (const float*)d_in[13];
  const float* bc   = (const float*)d_in[14];

  // workspace carve-up (256B aligned)
  size_t off = 0;
  auto alloc = [&](size_t bytes) -> char* {
    off = (off + 255) & ~(size_t)255;
    char* p = (char*)d_ws + off;
    off += bytes;
    return p;
  };
  bf16* Wih0_bf = (bf16*)alloc((size_t)GG * KP0 * 2);
  bf16* Whh0_bf = (bf16*)alloc((size_t)GG * HH * 2);
  bf16* Wih1_bf = (bf16*)alloc((size_t)GG * HH * 2);
  bf16* Whh1_bf = (bf16*)alloc((size_t)GG * HH * 2);
  bf16* Wphi_bf = (bf16*)alloc((size_t)MM * HH * 2);
  bf16* Wpsi_bf = (bf16*)alloc((size_t)MM * DD * 2);
  bf16* Wc_bf   = (bf16*)alloc((size_t)VV * (2 * HH) * 2);
  bf16* lis_bf  = (bf16*)alloc((size_t)BB * TT * DD * 2);
  bf16* x_bf    = (bf16*)alloc((size_t)BB * KP0 * 2);
  bf16* h0_bf   = (bf16*)alloc((size_t)BB * HH * 2);
  bf16* h1_bf   = (bf16*)alloc((size_t)BB * HH * 2);
  bf16* cat_bf  = (bf16*)alloc((size_t)BB * (2 * HH) * 2);
  float* comp_lis = (float*)alloc((size_t)BB * TT * MM * 4);
  float* gates    = (float*)alloc((size_t)BB * GG * 4);
  float* c0       = (float*)alloc((size_t)BB * HH * 4);
  float* c1       = (float*)alloc((size_t)BB * HH * 4);
  float* comp_dec = (float*)alloc((size_t)BB * MM * 4);
  float* logits   = (float*)alloc((size_t)BB * VV * 4);

  float* preds_out = (float*)d_out;                          // [100,B,V]
  float* attns_out = preds_out + (size_t)TDEC * BB * VV;     // [100,B,T]

  auto cvt = [&](const float* s, bf16* dst, int rows, int K, int Kp) {
    long long tot = (long long)rows * Kp;
    int blocks = (int)((tot + 255) / 256);
    cvt_pad_kernel<<<blocks, 256, 0, stream>>>(s, dst, rows, K, Kp);
  };
  // prologue: bf16 (K-padded) weight & activation copies
  cvt(Wih0, Wih0_bf, GG, VV + HH, KP0);
  cvt(Whh0, Whh0_bf, GG, HH, HH);
  cvt(Wih1, Wih1_bf, GG, HH, HH);
  cvt(Whh1, Whh1_bf, GG, HH, HH);
  cvt(Wphi, Wphi_bf, MM, HH, HH);
  cvt(Wpsi, Wpsi_bf, MM, DD, DD);
  cvt(Wc,   Wc_bf,   VV, 2 * HH, 2 * HH);
  cvt(listener, lis_bf, BB * TT, DD, DD);
  init_state_kernel<<<(BB * KP0) / 256, 256, 0, stream>>>(listener, x_bf, c0, c1, h0_bf, h1_bf);
  // comp_lis = relu(listener @ Wpsi.T + bpsi): M=8192, N=256, K=512
  wmma_gemm_dual<<<dim3(2, BB * TT / 32), 256, 0, stream>>>(
      lis_bf, DD, DD, nullptr, 0, 0, Wpsi_bf, nullptr, bpsi, nullptr, comp_lis, MM, 1);

  for (int step = 0; step < TDEC; ++step) {
    // LSTM layer 0 gates: x@Wih0.T + h0@Whh0.T + biases
    wmma_gemm_dual<<<dim3(16, 1), 256, 0, stream>>>(
        x_bf, KP0, KP0, h0_bf, HH, HH, Wih0_bf, Whh0_bf, bih0, bhh0, gates, GG, 0);
    lstm_cell_kernel<<<(BB * HH) / 256, 256, 0, stream>>>(gates, c0, h0_bf, nullptr);
    // LSTM layer 1 gates: h0n@Wih1.T + h1@Whh1.T + biases
    wmma_gemm_dual<<<dim3(16, 1), 256, 0, stream>>>(
        h0_bf, HH, HH, h1_bf, HH, HH, Wih1_bf, Whh1_bf, bih1, bhh1, gates, GG, 0);
    lstm_cell_kernel<<<(BB * HH) / 256, 256, 0, stream>>>(gates, c1, h1_bf, cat_bf);
    // phi: comp_dec = relu(h1n @ Wphi.T + bphi)
    wmma_gemm_dual<<<dim3(2, 1), 256, 0, stream>>>(
        h1_bf, HH, HH, nullptr, 0, 0, Wphi_bf, nullptr, bphi, nullptr, comp_dec, MM, 1);
    // energy -> softmax -> context (+ feedback context write)
    attend_kernel<<<BB, 256, 0, stream>>>(
        comp_dec, comp_lis, listener, attns_out + (size_t)step * BB * TT, cat_bf, x_bf);
    // logits = [rnn_out, context] @ Wc.T + bc   (125 n-tiles -> 16 blocks, guarded)
    wmma_gemm_dual<<<dim3(16, 1), 256, 0, stream>>>(
        cat_bf, 2 * HH, 2 * HH, nullptr, 0, 0, Wc_bf, nullptr, bc, nullptr, logits, VV, 0);
    // log_softmax -> preds output + feedback raw_pred write
    logsoftmax_kernel<<<BB, 256, 0, stream>>>(
        logits, preds_out + (size_t)step * BB * VV, x_bf);
  }
}
